// OutputAttention_74225624809729
// MI455X (gfx1250) — matrix operand order repaired and checked by reading, not compiled, not run on hardware
//
#include <hip/hip_runtime.h>
#include <math.h>

typedef __attribute__((ext_vector_type(16))) __bf16 v16bf;
typedef __attribute__((ext_vector_type(8)))  float  v8f;

#define B_ 32
#define S_ 2048
#define D_ 512
#define U_ 512

// round-to-nearest-even f32 -> bf16, packed pair into one u32
__device__ __forceinline__ unsigned pack_bf16(float lo, float hi) {
  unsigned a = __float_as_uint(lo), b = __float_as_uint(hi);
  a = (a + 0x7fffu + ((a >> 16) & 1u)) >> 16;
  b = (b + 0x7fffu + ((b >> 16) & 1u)) & 0xffff0000u;
  return a | b;
}

// ---------------------------------------------------------------------------
// Kernel 1: pack W1 [512x512] f32 -> bf16 B-matrix fragments for
// v_wmma_f32_16x16x32_bf16. Tile (nt,kt) = 32(K) x 16(N). Per-lane 8 u32:
// lanes 0-15: N=lane,    u32 j<4 = {K=2j, 2j+1},   j>=4 = {K=16+2(j-4), 17+2(j-4)}
// lanes16-31: N=lane-16, u32 j<4 = {K=8+2j, 9+2j}, j>=4 = {K=24+2(j-4), 25+2(j-4)}
// dest index = ((nt*16+kt)*32 + lane)*8 + j  == tid
// ---------------------------------------------------------------------------
__global__ void k_pack_w1(const float* __restrict__ W1, unsigned* __restrict__ w1p) {
  int tid  = blockIdx.x * 256 + threadIdx.x;   // 131072 total
  int j    = tid & 7;
  int lane = (tid >> 3) & 31;
  int kt   = (tid >> 8) & 15;
  int nt   = tid >> 12;
  int k0   = kt * 32 + (lane >> 4) * 8 + (j & 3) * 2 + (j >> 2) * 16;
  int u    = nt * 16 + (lane & 15);
  w1p[tid] = pack_bf16(W1[k0 * U_ + u], W1[(k0 + 1) * U_ + u]);
}

// ---------------------------------------------------------------------------
// Kernel 2: h_proj[b][u] = hidden[b] . W2[:,u]   (f32, tiny)
// ---------------------------------------------------------------------------
__global__ void k_hproj(const float* __restrict__ hidden, const float* __restrict__ W2,
                        float* __restrict__ hp) {
  int idx = blockIdx.x * 256 + threadIdx.x;    // 16384
  int b = idx >> 9, u = idx & 511;
  const float* h = hidden + b * D_;
  float acc = 0.f;
#pragma unroll 4
  for (int d = 0; d < D_; ++d) acc += h[d] * W2[d * U_ + u];
  hp[idx] = acc;
}

// ---------------------------------------------------------------------------
// Kernel 3: scores[b][s] = Vw . tanh(passage[b,s] @ W1 + h_proj[b]) + Vb
// 4 waves / WG, 64 passage rows per WG (one 16-row M-tile per wave).
// W1 B-fragments staged cooperatively into double-buffered LDS and shared by
// all 4 waves (4x less L2 traffic). A-fragments live in 128 VGPRs; B LDS
// reads are software-pipelined one k-step ahead of each WMMA.
// ---------------------------------------------------------------------------
__launch_bounds__(128)
__global__ void k_scores(const float* __restrict__ passage, const uint4* __restrict__ w1p4,
                         const float* __restrict__ hproj,   const float* __restrict__ Vw,
                         const float* __restrict__ Vb,      float* __restrict__ scores) {
  __shared__ unsigned ldsA[64 * 264];   // 64 rows x 512 bf16, padded stride (67.6 KB)
  __shared__ uint4    ldsB[2][1024];    // double buffer, 16 KB per N-tile

  int wg   = blockIdx.x;               // 1024 = B * S/64
  int b    = wg >> 5;
  int s0   = (wg & 31) * 64;
  int tid  = threadIdx.x;
  int wave = tid >> 5;
  int lane = tid & 31;

  // ---- stage A: 64x512 f32 -> bf16 in LDS (coalesced float4 loads) ----
  const float* src = passage + (size_t)(b * S_ + s0) * D_;
#pragma unroll 4
  for (int i = 0; i < 64; ++i) {
    int f4  = i * 128 + tid;           // 8192 float4's
    int row = f4 >> 7;
    int c4  = (f4 & 127) * 4;
    float4 v = *(const float4*)(src + row * D_ + c4);
    uint2 p;
    p.x = pack_bf16(v.x, v.y);
    p.y = pack_bf16(v.z, v.w);
    *(uint2*)&ldsA[row * 264 + (c4 >> 1)] = p;
  }

  // ---- stage B fragments for nt=0 ----
#pragma unroll
  for (int i = 0; i < 8; ++i)
    ldsB[0][i * 128 + tid] = w1p4[i * 128 + tid];
  __syncthreads();

  // ---- gather this wave's A fragments into registers (128 VGPRs) ----
  // ISA 16-bit A 16x32 layout per k-tile:
  // lanes 0-15 : row M=lane,    K 0..7  (bytes 0..15)  + K16..23 (bytes 32..47)
  // lanes16-31 : row M=lane-16, K 8..15 (bytes 16..31) + K24..31 (bytes 48..63)
  int arow16 = lane & 15;
  int hi4    = (lane >> 4) * 4;
  int abase  = (wave * 16 + arow16) * 264;
  uint4 Aq[16][2];
#pragma unroll
  for (int kt = 0; kt < 16; ++kt) {
    const unsigned* ap = &ldsA[abase + kt * 16 + hi4];
    Aq[kt][0] = *(const uint4*)ap;
    Aq[kt][1] = *(const uint4*)(ap + 8);
  }

  float sp[8] = {0.f,0.f,0.f,0.f,0.f,0.f,0.f,0.f};
  float vb0 = Vb[0];
  int bidx = lane * 2;                 // this lane's uint4 pair within a k-tile

  for (int nt = 0; nt < 32; ++nt) {
    int cur = nt & 1;

    // issue global loads for nt+1 into registers (hidden behind compute)
    uint4 nb[8];
    if (nt + 1 < 32) {
#pragma unroll
      for (int i = 0; i < 8; ++i)
        nb[i] = w1p4[(size_t)(nt + 1) * 1024 + i * 128 + tid];
      if (nt + 2 < 32)   // warm nearer caches for nt+2 (gfx1250 global_prefetch_b8)
        __builtin_prefetch((const char*)w1p4 + (size_t)(nt + 2) * 16384 + tid * 128, 0, 3);
    }

    // ---- 16x16 x K=512 block: 16 chained WMMAs, A from regs ----
    // B LDS reads pipelined one k-step ahead: load kt+1 before WMMA kt.
    v8f c = {};
    union { uint4 q[2]; v16bf v; } Bq[2];
    Bq[0].q[0] = ldsB[cur][bidx];
    Bq[0].q[1] = ldsB[cur][bidx + 1];
#pragma unroll
    for (int kt = 0; kt < 16; ++kt) {
      int pb = kt & 1;
      if (kt + 1 < 16) {
        Bq[pb ^ 1].q[0] = ldsB[cur][(kt + 1) * 64 + bidx];
        Bq[pb ^ 1].q[1] = ldsB[cur][(kt + 1) * 64 + bidx + 1];
      }
      union { uint4 q[2]; v16bf v; } A;
      A.q[0] = Aq[kt][0];
      A.q[1] = Aq[kt][1];
      c = __builtin_amdgcn_wmma_f32_16x16x32_bf16(false, A.v, false, Bq[pb].v,
                                                  (short)0, c, false, false);
    }

    // C layout: VGPR r, lanes 0-15 -> (M=r, N=lane); lanes16-31 -> (M=8+r, N=lane-16)
    float hp = hproj[b * U_ + nt * 16 + arow16];
    float vw = Vw[nt * 16 + arow16];
#pragma unroll
    for (int r = 0; r < 8; ++r) sp[r] += tanhf(c[r] + hp) * vw;

    // commit staged B for nt+1 into the other buffer
    if (nt + 1 < 32) {
#pragma unroll
      for (int i = 0; i < 8; ++i)
        ldsB[cur ^ 1][i * 128 + tid] = nb[i];
    }
    __syncthreads();
  }

  // reduce over N (16-lane halves); xor masks < 16 stay within each half
#pragma unroll
  for (int r = 0; r < 8; ++r) {
    float v = sp[r];
    v += __shfl_xor(v, 1, 32);
    v += __shfl_xor(v, 2, 32);
    v += __shfl_xor(v, 4, 32);
    v += __shfl_xor(v, 8, 32);
    sp[r] = v;
  }
  if ((lane & 15) == 0) {
    int moff = (lane >> 4) * 8;
#pragma unroll
    for (int r = 0; r < 8; ++r)
      scores[b * S_ + s0 + wave * 16 + moff + r] = sp[r] + vb0;
  }
}

// ---------------------------------------------------------------------------
// Kernel 4: masked softmax over S per batch -> attention weights (d_out region)
// ---------------------------------------------------------------------------
__global__ void k_softmax(const float* __restrict__ scores, const float* __restrict__ mask,
                          float* __restrict__ attn) {
  int b = blockIdx.x, tid = threadIdx.x;
  __shared__ float red[256];
  float loc[8];
  float mx = -3.4e38f;
#pragma unroll
  for (int i = 0; i < 8; ++i) {
    int s = tid + i * 256;
    float x = scores[b * S_ + s] + (1.f - mask[b * S_ + s]) * -1e30f;
    loc[i] = x;
    mx = fmaxf(mx, x);
  }
  red[tid] = mx; __syncthreads();
  for (int o = 128; o > 0; o >>= 1) {
    if (tid < o) red[tid] = fmaxf(red[tid], red[tid + o]);
    __syncthreads();
  }
  mx = red[0]; __syncthreads();
  float sum = 0.f;
#pragma unroll
  for (int i = 0; i < 8; ++i) { loc[i] = expf(loc[i] - mx); sum += loc[i]; }
  red[tid] = sum; __syncthreads();
  for (int o = 128; o > 0; o >>= 1) {
    if (tid < o) red[tid] += red[tid + o];
    __syncthreads();
  }
  float inv = 1.f / red[0];
#pragma unroll
  for (int i = 0; i < 8; ++i) attn[b * S_ + tid + i * 256] = loc[i] * inv;
}

// ---------------------------------------------------------------------------
// Kernels 5+6: context[b] = sum_s w[b,s] * passage[b,s,:]
// deterministic two-stage reduction over 16 S-chunks (no float atomics)
// ---------------------------------------------------------------------------
__global__ void k_ctx_partial(const float* __restrict__ passage, const float* __restrict__ attn,
                              float* __restrict__ part) {
  int bx = blockIdx.x;               // 1024 = 32 * 2 * 16
  int b = bx >> 5, dchunk = (bx >> 4) & 1, sc = bx & 15;
  int d = dchunk * 256 + threadIdx.x;
  const float* p = passage + ((size_t)b * S_ + sc * 128) * D_ + d;
  const float* w = attn + b * S_ + sc * 128;
  float acc = 0.f;
#pragma unroll 8
  for (int s = 0; s < 128; ++s) acc += w[s] * p[(size_t)s * D_];
  part[(b * 16 + sc) * D_ + d] = acc;
}

__global__ void k_ctx_final(const float* __restrict__ part, float* __restrict__ ctx) {
  int idx = blockIdx.x * 256 + threadIdx.x;   // 16384
  int b = idx >> 9, d = idx & 511;
  float acc = 0.f;
#pragma unroll
  for (int c = 0; c < 16; ++c) acc += part[(b * 16 + c) * D_ + d];
  ctx[idx] = acc;
}

// ---------------------------------------------------------------------------
extern "C" void kernel_launch(void* const* d_in, const int* in_sizes, int n_in,
                              void* d_out, int out_size, void* d_ws, size_t ws_size,
                              hipStream_t stream) {
  (void)in_sizes; (void)n_in; (void)out_size; (void)ws_size;
  const float* passage = (const float*)d_in[0];
  const float* hidden  = (const float*)d_in[1];
  const float* mask    = (const float*)d_in[2];
  const float* W1      = (const float*)d_in[3];
  const float* W2      = (const float*)d_in[4];
  const float* Vw      = (const float*)d_in[5];
  const float* Vb      = (const float*)d_in[6];

  float* out  = (float*)d_out;
  float* ctx  = out;               // [B, D]
  float* attn = out + B_ * D_;     // [B, S]

  char* ws = (char*)d_ws;
  unsigned* w1p    = (unsigned*)(ws);            // 512 KB packed bf16 W1 fragments
  float*    hproj  = (float*)(ws + 524288);      //  64 KB
  float*    scores = (float*)(ws + 589824);      // 256 KB
  float*    part   = (float*)(ws + 851968);      //   1 MB

  k_pack_w1    <<<512,  256, 0, stream>>>(W1, w1p);
  k_hproj      <<<64,   256, 0, stream>>>(hidden, W2, hproj);
  k_scores     <<<1024, 128, 0, stream>>>(passage, (const uint4*)w1p, hproj, Vw, Vb, scores);
  k_softmax    <<<32,   256, 0, stream>>>(scores, mask, attn);
  k_ctx_partial<<<1024, 256, 0, stream>>>(passage, attn, part);
  k_ctx_final  <<<64,   256, 0, stream>>>(part, ctx);
}
